// MyLSTM_34462817583397
// MI455X (gfx1250) — matrix operand order repaired and checked by reading, not compiled, not run on hardware
//
#include <hip/hip_runtime.h>
#include <math.h>

// ---------------------------------------------------------------------------
// LSTM on MI455X (gfx1250): bf16 WMMA GEMMs, fused gate epilogue.
//   x = embed[text]                      (gather -> bf16, K padded 300->320)
//   xU = x @ [Ui|Uf|Uc|Uo]               (WMMA GEMM, f32 out, 512 MB stream)
//   512x: gates = xU_t + h @ V ; LSTM cell update
//         (WMMA + double-buffered async V staging in LDS + fused epilogue)
//   out = h_n @ W + b                    (tiny scalar kernel)
// ---------------------------------------------------------------------------

typedef __attribute__((ext_vector_type(16))) __bf16 v16bf;
typedef __attribute__((ext_vector_type(8)))  __bf16 v8bf;
typedef __attribute__((ext_vector_type(8)))  float  v8f;
typedef int v4i __attribute__((ext_vector_type(4)));

#define BSZ   64
#define SEQ   512
#define EMB   300
#define EPAD  320          // K padded to multiple of 32
#define HID   1024
#define H4    4096
#define POL   3

// ---- workspace layout (bytes) ----
#define OFF_XPAD  0u                                   // [32768][320] bf16 = 20,971,520
#define OFF_USW   (OFF_XPAD + 32768u*EPAD*2u)          // [10][256][32][16] bf16 = 2,621,440
#define OFF_VSW   (OFF_USW  + (EPAD/32u)*256u*512u*2u) // [32][256][32][16] bf16 = 8,388,608
#define OFF_XU    (OFF_VSW  + (HID/32u)*256u*512u*2u)  // [512][64][4096] f32 = 536,870,912
#define OFF_C     (OFF_XU   + (size_t)SEQ*BSZ*H4*4u)   // [64][1024] f32
#define OFF_HF    (OFF_C    + (size_t)BSZ*HID*4u)      // [64][1024] f32
#define OFF_HB0   (OFF_HF   + (size_t)BSZ*HID*4u)      // [64][1024] bf16
#define OFF_HB1   (OFF_HB0  + (size_t)BSZ*HID*2u)      // [64][1024] bf16

// Async global->LDS copy (CDNA5): signature discovered via compile probe:
//   (v4i addrspace(1)* gsrc, v4i addrspace(3)* ldsdst, int offset, int cpol)
#if __has_builtin(__builtin_amdgcn_global_load_async_to_lds_b128)
#define HAVE_ASYNC_LDS 1
typedef __attribute__((address_space(1))) v4i g_v4i;
typedef __attribute__((address_space(3))) v4i l_v4i;
#define ASYNC_CP16(gsrc, ldst) \
  __builtin_amdgcn_global_load_async_to_lds_b128((g_v4i*)(gsrc), (l_v4i*)(ldst), 0, 0)
#define WAIT_ASYNC_2() asm volatile("s_wait_asynccnt 0x2" ::: "memory")
#define WAIT_ASYNC_0() asm volatile("s_wait_asynccnt 0x0" ::: "memory")
#else
#define HAVE_ASYNC_LDS 0
#endif

// ---------------------------------------------------------------------------
// Embedding gather -> bf16, row r = s*64 + b (matches xU's [S,B,4H] order),
// K padded with zeros from 300..319.
// ---------------------------------------------------------------------------
__global__ __launch_bounds__(256)
void k_embed(const int* __restrict__ text, const float* __restrict__ tab,
             __bf16* __restrict__ xpad) {
  long id = (long)blockIdx.x * blockDim.x + threadIdx.x;
  if (id >= (long)BSZ * SEQ * EPAD) return;
  int  e = (int)(id % EPAD);
  long r = id / EPAD;
  int  b = (int)(r % BSZ);
  int  s = (int)(r / BSZ);
  float v = 0.0f;
  if (e < EMB) {
    int tok = text[b * SEQ + s];
    v = tab[(long)tok * EMB + e];
  }
  xpad[id] = (__bf16)v;
}

// ---------------------------------------------------------------------------
// Swizzle the 4 gate weight blocks [Krows,1024] each (cols i|f|c|o concat to
// N=4096) into fragment-contiguous bf16 B-layout:
//   out[((kt*256 + ntile)*32 + lane)*16 + idx]
//   lane: col = lane&15 ; idx: K = (idx>>3)*16 + (lane>>4)*8 + (idx&7)
// ---------------------------------------------------------------------------
__global__ __launch_bounds__(256)
void k_swzB(const float* __restrict__ S0, const float* __restrict__ S1,
            const float* __restrict__ S2, const float* __restrict__ S3,
            __bf16* __restrict__ out, int ktiles, int krows) {
  long id = (long)blockIdx.x * blockDim.x + threadIdx.x;
  if (id >= (long)ktiles * 256 * 512) return;
  int idx  = (int)(id & 15);
  int lane = (int)((id >> 4) & 31);
  long rest = id >> 9;
  int ntg = (int)(rest & 255);
  int kt  = (int)(rest >> 8);
  int K = kt * 32 + ((idx >> 3) << 4) + ((lane >> 4) << 3) + (idx & 7);
  int N = (ntg << 4) + (lane & 15);
  int g = N >> 10;
  int j = N & 1023;
  const float* Sg = (g == 0) ? S0 : (g == 1) ? S1 : (g == 2) ? S2 : S3;
  float v = (K < krows) ? Sg[(long)K * HID + j] : 0.0f;
  out[id] = (__bf16)v;
}

__device__ __forceinline__ v16bf load_a_frag(const __bf16* p) {
  v8bf lo = *(const v8bf*)(p);        // K run [hb .. hb+7]
  v8bf hi = *(const v8bf*)(p + 16);   // K run [16+hb .. 16+hb+7]
  v16bf a;
#pragma unroll
  for (int i = 0; i < 8; ++i) { a[i] = lo[i]; a[8 + i] = hi[i]; }
  return a;
}

// ---------------------------------------------------------------------------
// xU = xpad @ Usw  : [32768,320] x [320,4096] -> f32 [32768,4096]
// grid (64, 512) x 128 threads; wave = 16 rows x 64 cols (4 wmma tiles).
// 32768 blocks -> occupancy hides memory latency here.
// ---------------------------------------------------------------------------
__global__ __launch_bounds__(128)
void k_xu_gemm(const __bf16* __restrict__ xpad, const __bf16* __restrict__ Usw,
               float* __restrict__ xU) {
  int lane = threadIdx.x & 31;
  int wave = threadIdx.x >> 5;
  int rbase = blockIdx.y * 64 + wave * 16;
  int nblk  = blockIdx.x;                 // 64-column group
  int hb    = (lane >> 4) << 3;           // 0 or 8
  int arow  = rbase + (lane & 15);

  v8f a0 = {}, a1 = {}, a2 = {}, a3 = {};
  const __bf16* ap = xpad + (long)arow * EPAD + hb;

  for (int kt = 0; kt < EPAD / 32; ++kt) {
    v16bf a = load_a_frag(ap);
    ap += 32;
    const __bf16* bp = Usw + (((long)kt * 256 + nblk * 4) * 32 + lane) * 16;
    v16bf b0 = *(const v16bf*)(bp);
    v16bf b1 = *(const v16bf*)(bp + 512);
    v16bf b2 = *(const v16bf*)(bp + 1024);
    v16bf b3 = *(const v16bf*)(bp + 1536);
    a0 = __builtin_amdgcn_wmma_f32_16x16x32_bf16(false, a, false, b0, (short)0, a0, false, false);
    a1 = __builtin_amdgcn_wmma_f32_16x16x32_bf16(false, a, false, b1, (short)0, a1, false, false);
    a2 = __builtin_amdgcn_wmma_f32_16x16x32_bf16(false, a, false, b2, (short)0, a2, false, false);
    a3 = __builtin_amdgcn_wmma_f32_16x16x32_bf16(false, a, false, b3, (short)0, a3, false, false);
  }

  int rowb = rbase + ((lane >> 4) << 3);
  int col0 = nblk * 64 + (lane & 15);
#pragma unroll
  for (int i = 0; i < 8; ++i) {
    float* o = xU + (long)(rowb + i) * H4 + col0;
    o[0] = a0[i]; o[16] = a1[i]; o[32] = a2[i]; o[48] = a3[i];
  }
}

// ---------------------------------------------------------------------------
// Zero the recurrent state.
// ---------------------------------------------------------------------------
__global__ __launch_bounds__(256)
void k_init(float* __restrict__ c, float* __restrict__ hf,
            __bf16* __restrict__ h0, __bf16* __restrict__ h1) {
  int i = blockIdx.x * blockDim.x + threadIdx.x;
  if (i < BSZ * HID) {
    c[i] = 0.0f; hf[i] = 0.0f;
    h0[i] = (__bf16)0.0f; h1[i] = (__bf16)0.0f;
  }
}

// ---------------------------------------------------------------------------
// One LSTM step:  gates[64,4096] = xU_s + h @ V ; cell update (fused).
// grid 32 blocks x 256 threads (8 waves). Wave w: rows (w&3)*16,
// j-group jg = blockIdx.x*2 + (w>>2). The block's 8 B-fragments per k-tile
// (2 jg x 4 gates, 8 KB) are staged in LDS, double-buffered: stage kt+1
// asynchronously while each wave's 4 WMMAs of kt run out of LDS. h is register
// double-buffered the same way.
// ---------------------------------------------------------------------------
__global__ __launch_bounds__(256)
void k_lstm_step(const float* __restrict__ xU_s, const __bf16* __restrict__ Vsw,
                 const __bf16* __restrict__ hin, __bf16* __restrict__ hout,
                 float* __restrict__ hf, float* __restrict__ c,
                 const float* __restrict__ b_i, const float* __restrict__ b_f,
                 const float* __restrict__ b_c, const float* __restrict__ b_o) {
  __shared__ __bf16 smem[2 * 8 * 32 * 16];     // 2 bufs x 8 frags x 1 KB

  int tid   = threadIdx.x;
  int lane  = tid & 31;
  int wave  = tid >> 5;
  int rgrp  = wave & 3;                        // row group (16 rows)
  int jgsel = wave >> 2;                       // 0/1 within block
  int jg    = blockIdx.x * 2 + jgsel;          // 16-col group within a gate
  int hb    = (lane >> 4) << 3;
  int arow  = rgrp * 16 + (lane & 15);

  // cooperative stage mapping: thread t loads fragment f = t>>5, lane l = t&31
  int sf = tid >> 5;                           // 0..7 : f = js*4 + g
  int sg = sf & 3;
  int sjs = sf >> 2;
  int sntg = sg * 64 + blockIdx.x * 2 + sjs;
  const __bf16* sbase = Vsw + ((long)sntg * 32 + lane) * 16;   // + kt*256*512
  __bf16* sdst0 = smem + (sf * 32 + lane) * 16;                // + buf*8*512

  v8f acc[4] = {v8f{}, v8f{}, v8f{}, v8f{}};
  const __bf16* ap = hin + (long)arow * HID + hb;

  // ---- stage kt = 0 into buffer 0, preload first A fragment ----
#if HAVE_ASYNC_LDS
  ASYNC_CP16(sbase,     sdst0);
  ASYNC_CP16(sbase + 8, sdst0 + 8);
#else
  { v16bf t = *(const v16bf*)sbase; *(v16bf*)sdst0 = t; }
#endif
  v16bf a_cur = load_a_frag(ap);

  for (int kt = 0; kt < HID / 32; ++kt) {
    int cur = kt & 1;
    v16bf a_next;
    if (kt + 1 < HID / 32) {
      // issue next stage (async) + next A fragment while kt computes
      const __bf16* s = sbase + (long)(kt + 1) * 256 * 512;
      __bf16* d = smem + (cur ^ 1) * 8 * 512 + (sf * 32 + lane) * 16;
#if HAVE_ASYNC_LDS
      ASYNC_CP16(s,     d);
      ASYNC_CP16(s + 8, d + 8);
#else
      { v16bf t = *(const v16bf*)s; *(v16bf*)d = t; }
#endif
      a_next = load_a_frag(ap + (long)(kt + 1) * 32);
    }
#if HAVE_ASYNC_LDS
    if (kt + 1 < HID / 32) { WAIT_ASYNC_2(); } else { WAIT_ASYNC_0(); }
#endif
    __syncthreads();                           // stage kt visible to all waves

    const __bf16* bbuf = smem + cur * 8 * 512 + ((jgsel * 4) * 32 + lane) * 16;
#pragma unroll
    for (int g = 0; g < 4; ++g) {
      v16bf b = *(const v16bf*)(bbuf + g * 512);   // ds_load_b128 x2
      acc[g] = __builtin_amdgcn_wmma_f32_16x16x32_bf16(
          false, a_cur, false, b, (short)0, acc[g], false, false);
    }
    __syncthreads();                           // all waves done with buf cur
    a_cur = a_next;
  }

  int rowb = rgrp * 16 + ((lane >> 4) << 3);
  int j    = jg * 16 + (lane & 15);
#pragma unroll
  for (int i = 0; i < 8; ++i) {
    int row = rowb + i;
    long off = (long)row * H4 + j;
    float xi = acc[0][i] + xU_s[off]        + b_i[j];
    float xf = acc[1][i] + xU_s[off + 1024] + b_f[j];
    float xg = acc[2][i] + xU_s[off + 2048] + b_c[j];
    float xo = acc[3][i] + xU_s[off + 3072] + b_o[j];
    float iv = 1.0f / (1.0f + expf(-xi));
    float fv = 1.0f / (1.0f + expf(-xf));
    float gv = tanhf(xg);
    float ov = 1.0f / (1.0f + expf(-xo));
    long hoff = (long)row * HID + j;
    float cn = fv * c[hoff] + iv * gv;
    float hn = ov * tanhf(cn);
    c[hoff]  = cn;
    hf[hoff] = hn;
    hout[hoff] = (__bf16)hn;
  }
}

// ---------------------------------------------------------------------------
// out[64,3] = h_n @ W + b  (tiny; 192 dot products of length 1024)
// ---------------------------------------------------------------------------
__global__ __launch_bounds__(192)
void k_dense(const float* __restrict__ hf, const float* __restrict__ W,
             const float* __restrict__ bd, float* __restrict__ out) {
  int t = threadIdx.x;
  if (t >= BSZ * POL) return;
  int b = t / POL, p = t % POL;
  float acc = bd[p];
  for (int k = 0; k < HID; ++k) acc += hf[b * HID + k] * W[k * POL + p];
  out[t] = acc;
}

// ---------------------------------------------------------------------------
extern "C" void kernel_launch(void* const* d_in, const int* in_sizes, int n_in,
                              void* d_out, int out_size, void* d_ws, size_t ws_size,
                              hipStream_t stream) {
  (void)in_sizes; (void)n_in; (void)out_size; (void)ws_size;
  const int*   text = (const int*)  d_in[0];
  const float* tab  = (const float*)d_in[1];
  const float* U_i  = (const float*)d_in[2];
  const float* U_f  = (const float*)d_in[3];
  const float* U_c  = (const float*)d_in[4];
  const float* U_o  = (const float*)d_in[5];
  const float* V_i  = (const float*)d_in[6];
  const float* V_f  = (const float*)d_in[7];
  const float* V_c  = (const float*)d_in[8];
  const float* V_o  = (const float*)d_in[9];
  const float* b_i  = (const float*)d_in[10];
  const float* b_f  = (const float*)d_in[11];
  const float* b_c  = (const float*)d_in[12];
  const float* b_o  = (const float*)d_in[13];
  const float* Wd   = (const float*)d_in[14];
  const float* bd   = (const float*)d_in[15];

  char* w = (char*)d_ws;
  __bf16* xpad = (__bf16*)(w + OFF_XPAD);
  __bf16* Usw  = (__bf16*)(w + OFF_USW);
  __bf16* Vsw  = (__bf16*)(w + OFF_VSW);
  float*  xU   = (float*) (w + OFF_XU);
  float*  c    = (float*) (w + OFF_C);
  float*  hf   = (float*) (w + OFF_HF);
  __bf16* hb0  = (__bf16*)(w + OFF_HB0);
  __bf16* hb1  = (__bf16*)(w + OFF_HB1);

  // 1. gather + pad + bf16
  {
    long n = (long)BSZ * SEQ * EPAD;
    k_embed<<<(unsigned)((n + 255) / 256), 256, 0, stream>>>(text, tab, xpad);
  }
  // 2. weight swizzles (B-fragment layout)
  k_swzB<<<((EPAD / 32) * 256 * 512 + 255) / 256, 256, 0, stream>>>(
      U_i, U_f, U_c, U_o, Usw, EPAD / 32, EMB);
  k_swzB<<<((HID / 32) * 256 * 512 + 255) / 256, 256, 0, stream>>>(
      V_i, V_f, V_c, V_o, Vsw, HID / 32, HID);
  // 3. input projection GEMM (WMMA)
  k_xu_gemm<<<dim3(64, 512), 128, 0, stream>>>(xpad, Usw, xU);
  // 4. state init
  k_init<<<(BSZ * HID + 255) / 256, 256, 0, stream>>>(c, hf, hb0, hb1);
  // 5. sequential scan: 512 fused WMMA + LSTM-cell steps, h ping-pong
  for (int s = 0; s < SEQ; ++s) {
    const __bf16* hin  = (s & 1) ? hb1 : hb0;
    __bf16*       hout = (s & 1) ? hb0 : hb1;
    k_lstm_step<<<32, 256, 0, stream>>>(xU + (size_t)s * BSZ * H4, Vsw,
                                        hin, hout, hf, c, b_i, b_f, b_c, b_o);
  }
  // 6. policy head
  k_dense<<<1, 192, 0, stream>>>(hf, Wd, bd, (float*)d_out);
}
